// WindowedAttention_67190468379038
// MI455X (gfx1250) — matrix operand order repaired and checked by reading, not compiled, not run on hardware
//
#include <hip/hip_runtime.h>
#include <hip/hip_bf16.h>

// ---------------------------------------------------------------------------
// Windowed attention, MI455X (gfx1250, wave32, WMMA + async-LDS + TDM).
//   S=4096, E=1024, H=16, hd=64, WS=512, LB=LF=1 (3 chunks of 512 keys).
// Pipeline:
//   1) one-shot f32->f16 conversion of x and all 4 weight matrices
//   2) 3x QKV proj GEMM: f16 WMMA, f32 accum, double-buffered LDS tiles
//      staged with GLOBAL_LOAD_ASYNC_TO_LDS_B128 (ASYNCcnt) when available
//   3) flash windowed attention; K chunks staged by TENSOR_LOAD_TO_LDS
//      (TENSORcnt, pad_enable reproduces the padded LDS pitch) when available
//   4) output proj GEMM (f32 out)
// Workspace: x16(8MB) + w16 x4 (8MB) + Qh/Kh/Vh(24MB) + ctx(8MB) = 48 MB.
// ---------------------------------------------------------------------------

#define S_LEN 4096
#define E_DIM 1024
#define HEADS 16
#define HD 64
#define WSQ 512
#define NWIN 8
#define NCHUNK 24          // 3*512/64 key sub-chunks per window

typedef __attribute__((ext_vector_type(16))) _Float16 v16h;
typedef __attribute__((ext_vector_type(8)))  _Float16 v8h;
typedef __attribute__((ext_vector_type(8)))  float    v8f;
typedef __attribute__((ext_vector_type(4)))  float    v4f;

union F16x16 { v16h v; v8h h[2]; };

// ---- CDNA5 feature gates (also cleanly disable these on the host pass) ----
#if __has_builtin(__builtin_amdgcn_global_load_async_to_lds_b128) && \
    __has_builtin(__builtin_amdgcn_s_wait_asynccnt)
#define USE_ASYNC_LDS 1
#else
#define USE_ASYNC_LDS 0
#endif

#if __has_builtin(__builtin_amdgcn_tensor_load_to_lds) && \
    __has_builtin(__builtin_amdgcn_s_wait_tensorcnt)
#define USE_TDM 1
#else
#define USE_TDM 0
#endif

__device__ __forceinline__ v8f wmma_f16(v16h a, v16h b, v8f c) {
    return __builtin_amdgcn_wmma_f32_16x16x32_f16(
        false, a, false, b, (short)0, c, false, false);
}

// A-fragment (16x32 f16): lanes 0-15 = rows, K 0..7 & 16..23; lanes 16-31: +8.
__device__ __forceinline__ v16h load_frag_a(const _Float16* base, int stride, int lane) {
    int row = lane & 15;
    int k0  = (lane >> 4) * 8;
    F16x16 f;
    f.h[0] = *(const v8h*)(base + row * stride + k0);
    f.h[1] = *(const v8h*)(base + row * stride + k0 + 16);
    return f.v;
}

// B-fragment (32x16 f16): lane = col N, lanes 0-15 K 0..7 & 16..23; lanes 16-31: +8.
// base is column-contiguous: element(k, n) = base[n*stride + k].
__device__ __forceinline__ v16h load_frag_b(const _Float16* base, int stride, int lane) {
    int col = lane & 15;
    int k0  = (lane >> 4) * 8;
    F16x16 f;
    f.h[0] = *(const v8h*)(base + col * stride + k0);
    f.h[1] = *(const v8h*)(base + col * stride + k0 + 16);
    return f.v;
}

#if USE_ASYNC_LDS
// Builtin prototype (from the round-2 diagnostic): first param is
// 'int __vector(4) __device__ *' i.e. GCC-vector int4 in AS1; dst is the
// AS3 counterpart. Cast through integers to the exact pointer types.
typedef int gvec4i __attribute__((vector_size(16)));
typedef __attribute__((address_space(1))) gvec4i* as1_v4i_p;
typedef __attribute__((address_space(3))) gvec4i* as3_v4i_p;
#endif

// 16-byte global->LDS copy: async on gfx1250 toolchains that expose it.
// AS1 value == generic value; low 32 bits of a generic LDS addr == LDS offset.
__device__ __forceinline__ void lds_copy16(const _Float16* gsrc, _Float16* ldst) {
#if USE_ASYNC_LDS
    __builtin_amdgcn_global_load_async_to_lds_b128(
        (as1_v4i_p)(unsigned long long)gsrc,
        (as3_v4i_p)(unsigned)(unsigned long long)ldst,
        0, 0);
#else
    *(v8h*)ldst = *(const v8h*)gsrc;
#endif
}

__device__ __forceinline__ void lds_copy_join() {
#if USE_ASYNC_LDS
    __builtin_amdgcn_s_wait_asynccnt(0);
#endif
}

#if USE_TDM
typedef unsigned int tdm_u32x4 __attribute__((ext_vector_type(4)));
typedef int          tdm_i32x8 __attribute__((ext_vector_type(8)));
typedef int          tdm_i32x4 __attribute__((ext_vector_type(4)));

// TDM: load a 64x64 f16 tile (row-major, row stride 64 elems) from global into
// LDS with pad_enable: pad 4 DWORDs (8 halves) after every 32 DWORDs (64-half
// row) -> matches the 72-half LDS pitch. D# fields per CDNA5 ISA §8.3/8.4.
__device__ __forceinline__ void tdm_load_k64x64(unsigned lds_off, const _Float16* gsrc) {
    unsigned long long ga = (unsigned long long)gsrc;
    tdm_u32x4 g0;
    g0.x = 1u;                                          // count=1, user descriptor
    g0.y = lds_off;                                     // lds_addr (bytes)
    g0.z = (unsigned)ga;                                // global_addr[31:0]
    g0.w = (unsigned)((ga >> 32) & 0x01FFFFFFu) | (2u << 30);  // addr[56:32]|type=2
    tdm_i32x8 g1;
    g1[0] = (1 << 16) | (1 << 20) | (4 << 22) | (3 << 25); // dsz=2B,pad_en,int=32dw,amt=4dw
    g1[1] = (64 << 16);                                 // tensor_dim0 = 64
    g1[2] = (64 << 16);                                 // tensor_dim1 = 64
    g1[3] = (64 << 16);                                 // tile_dim0 = 64
    g1[4] = 64;                                         // tile_dim1 = 64, tile_dim2 = 0
    g1[5] = 64;                                         // tensor_dim0_stride = 64
    g1[6] = 0;
    g1[7] = 0;
    tdm_i32x4 z4 = (tdm_i32x4)0;
#if __clang_major__ >= 23
    __builtin_amdgcn_tensor_load_to_lds(g0, g1, z4, z4, (tdm_i32x8)0, 0);
#else
    __builtin_amdgcn_tensor_load_to_lds(g0, g1, z4, z4, 0);
#endif
}
#endif // USE_TDM

// ---------------------------------------------------------------------------
// Kernel 0: one-shot f32 -> f16 conversion (n multiple of 8).
// ---------------------------------------------------------------------------
__global__ void __launch_bounds__(256)
cvt_f16_kernel(const float* __restrict__ src, _Float16* __restrict__ dst, int n)
{
    int i = (blockIdx.x * 256 + threadIdx.x) * 8;
    if (i >= n) return;
    v4f a = *(const v4f*)(src + i);
    v4f b = *(const v4f*)(src + i + 4);
    v8h o;
    #pragma unroll
    for (int j = 0; j < 4; ++j) { o[j] = (_Float16)a[j]; o[4 + j] = (_Float16)b[j]; }
    *(v8h*)(dst + i) = o;
}

// ---------------------------------------------------------------------------
// GEMM: y = A(f16)[M][1024] @ W(f16)[N][1024]^T + bias.
// MODE 0: store f16 head-major [H][S][64].  MODE 1: store f32 flat [S][E].
// grid(E/128, M/128), block 256 (8 waves), wave tile 64x32, double-buffered
// LDS tiles staged via async global->LDS copies.
// ---------------------------------------------------------------------------
template <int MODE>
__global__ void __launch_bounds__(256)
gemm_wmma_kernel(const _Float16* __restrict__ A, const _Float16* __restrict__ W,
                 const float* __restrict__ bias,
                 _Float16* __restrict__ outh, float* __restrict__ outf)
{
    __shared__ __align__(16) _Float16 As[2][128][40];   // 128 x 32 (pad 40)
    __shared__ __align__(16) _Float16 Bs[2][128][40];

    const int tid  = threadIdx.x;
    const int lane = tid & 31;
    const int wid  = tid >> 5;
    const int wm   = wid >> 2;
    const int wn   = wid & 3;
    const int m0   = blockIdx.y * 128;
    const int n0   = blockIdx.x * 128;

    v8f acc[4][2];
    for (int i = 0; i < 4; ++i) { acc[i][0] = (v8f)0.f; acc[i][1] = (v8f)0.f; }

    const int lrow = tid >> 1;
    const int lcb  = (tid & 1) * 16;
    const _Float16* ap = A + (size_t)(m0 + lrow) * E_DIM + lcb;
    const _Float16* wp = W + (size_t)(n0 + lrow) * E_DIM + lcb;

    auto stage = [&](int buf, int k0) {
        lds_copy16(ap + k0,     &As[buf][lrow][lcb]);
        lds_copy16(ap + k0 + 8, &As[buf][lrow][lcb + 8]);
        lds_copy16(wp + k0,     &Bs[buf][lrow][lcb]);
        lds_copy16(wp + k0 + 8, &Bs[buf][lrow][lcb + 8]);
    };

    stage(0, 0);
    lds_copy_join();
    __syncthreads();

    for (int k0 = 0; k0 < E_DIM; k0 += 32) {
        const int buf = (k0 >> 5) & 1;
        if (k0 + 32 < E_DIM) stage(buf ^ 1, k0 + 32);   // prefetch next tile

        v16h bf0 = load_frag_b(&Bs[buf][wn * 32][0],      40, lane);
        v16h bf1 = load_frag_b(&Bs[buf][wn * 32 + 16][0], 40, lane);
        #pragma unroll
        for (int mt = 0; mt < 4; ++mt) {
            v16h af = load_frag_a(&As[buf][wm * 64 + mt * 16][0], 40, lane);
            acc[mt][0] = wmma_f16(af, bf0, acc[mt][0]);
            acc[mt][1] = wmma_f16(af, bf1, acc[mt][1]);
        }
        lds_copy_join();
        __syncthreads();
    }

    const int colA = n0 + wn * 32 + (lane & 15);
    const float b0 = bias[colA], b1 = bias[colA + 16];
    #pragma unroll
    for (int mt = 0; mt < 4; ++mt) {
        #pragma unroll
        for (int i = 0; i < 8; ++i) {
            int s = m0 + wm * 64 + mt * 16 + i + 8 * (lane >> 4);
            float v0 = acc[mt][0][i] + b0;
            float v1 = acc[mt][1][i] + b1;
            if (MODE == 0) {
                int e0 = colA, e1 = colA + 16;
                outh[((size_t)(e0 >> 6) * S_LEN + s) * HD + (e0 & 63)] = (_Float16)v0;
                outh[((size_t)(e1 >> 6) * S_LEN + s) * HD + (e1 & 63)] = (_Float16)v1;
            } else {
                outf[(size_t)s * E_DIM + colA]      = v0;
                outf[(size_t)s * E_DIM + colA + 16] = v1;
            }
        }
    }
}

// ---------------------------------------------------------------------------
// Windowed flash attention. grid(8 qtiles, 8 windows, 16 heads), block 128
// (4 waves, 16 query rows each). Full in-range K chunks staged by TDM; the
// boundary chunks replicate the reference quirk: out-of-range keys get
// K=V=-1.0 and are NOT masked; in-range keys masked when |kt - q| > 512.
// ---------------------------------------------------------------------------
__global__ void __launch_bounds__(128)
attn_kernel(const _Float16* __restrict__ Qh, const _Float16* __restrict__ Kh,
            const _Float16* __restrict__ Vh, _Float16* __restrict__ ctx)
{
    __shared__ __align__(16) _Float16 Ks[64][72];      // K chunk, row-major
    __shared__ __align__(16) _Float16 Vt[64][72];      // V chunk, transposed [d][k]
    __shared__ __align__(16) _Float16 Ps[4][16][72];   // per-wave P scratch

    const int tid  = threadIdx.x;
    const int lane = tid & 31;
    const int wid  = tid >> 5;
    const int lrow = lane & 15;
    const int lhal = lane >> 4;
    const int h    = blockIdx.z;
    const int w    = blockIdx.y;
    const int qt   = blockIdx.x;
    const int q0   = w * WSQ + qt * 64 + wid * 16;

    const _Float16* qbase = Qh + ((size_t)h * S_LEN + q0) * HD;
    const v16h qa0 = load_frag_a(qbase,      HD, lane);
    const v16h qa1 = load_frag_a(qbase + 32, HD, lane);

    float rmax[8], rsum[8];
    v8f oacc[4];
    #pragma unroll
    for (int i = 0; i < 8; ++i) { rmax[i] = -1e30f; rsum[i] = 0.f; }
    #pragma unroll
    for (int t = 0; t < 4; ++t) oacc[t] = (v8f)0.f;

    const int kstart = w * WSQ - WSQ;
    const int srow   = tid >> 1;
    const int scb    = (tid & 1) * 32;
#if USE_TDM
    const unsigned ks_lds = (unsigned)(unsigned long long)&Ks[0][0];
#endif

    for (int c = 0; c < NCHUNK; ++c) {
        const int kb = kstart + c * 64;
        const bool full = (kb >= 0) && (kb + 64 <= S_LEN);
        (void)full;

        // ---- stage V transposed (always manual: transpose on the fly) ----
        {
            int kt = kb + srow;
            if (kt >= 0 && kt < S_LEN) {
                const _Float16* vp = Vh + ((size_t)h * S_LEN + kt) * HD + scb;
                #pragma unroll
                for (int i = 0; i < 32; i += 8) {
                    v8h vv = *(const v8h*)(vp + i);
                    #pragma unroll
                    for (int j = 0; j < 8; ++j) Vt[scb + i + j][srow] = vv[j];
                }
            } else {
                #pragma unroll
                for (int i = 0; i < 32; ++i) Vt[scb + i][srow] = (_Float16)(-1.0f);
            }
        }

        // ---- stage K: one TDM descriptor for full chunks, else per-row ----
#if USE_TDM
        if (full) {
            if (wid == 0)
                tdm_load_k64x64(ks_lds, Kh + ((size_t)h * S_LEN + kb) * HD);
        } else
#endif
        {
            int kt = kb + srow;
            if (kt >= 0 && kt < S_LEN) {
                const _Float16* kp = Kh + ((size_t)h * S_LEN + kt) * HD + scb;
                lds_copy16(kp,      &Ks[srow][scb]);
                lds_copy16(kp + 8,  &Ks[srow][scb + 8]);
                lds_copy16(kp + 16, &Ks[srow][scb + 16]);
                lds_copy16(kp + 24, &Ks[srow][scb + 24]);
            } else {
                #pragma unroll
                for (int i = 0; i < 32; ++i) Ks[srow][scb + i] = (_Float16)(-1.0f);
            }
        }
#if USE_TDM
        if (full && wid == 0) __builtin_amdgcn_s_wait_tensorcnt(0);
#endif
        lds_copy_join();
        __syncthreads();

        // ---- scores: 4 tiles of 16 keys, 2 WMMA (k=32 over d) each ----
        v8f sc[4];
        #pragma unroll
        for (int ns = 0; ns < 4; ++ns) {
            v16h bk0 = load_frag_b(&Ks[ns * 16][0],      72, lane);
            v16h bk1 = load_frag_b(&Ks[ns * 16][0] + 32, 72, lane);
            v8f z = (v8f)0.f;
            z      = wmma_f16(qa0, bk0, z);
            sc[ns] = wmma_f16(qa1, bk1, z);
        }

        // ---- mask + online softmax (C layout: row = i+8*lhal, col = lrow) --
        float nmax[8];
        #pragma unroll
        for (int i = 0; i < 8; ++i) nmax[i] = rmax[i];
        #pragma unroll
        for (int ns = 0; ns < 4; ++ns) {
            int kcol = kb + ns * 16 + lrow;
            bool inr = (kcol >= 0) && (kcol < S_LEN);
            #pragma unroll
            for (int i = 0; i < 8; ++i) {
                int q = q0 + i + 8 * lhal;
                bool over = inr && ((kcol > q + WSQ) || (kcol < q - WSQ));
                float s = over ? -1e30f : sc[ns][i];
                sc[ns][i] = s;
                nmax[i] = fmaxf(nmax[i], s);
            }
        }
        #pragma unroll
        for (int i = 0; i < 8; ++i) {
            float m = nmax[i];
            m = fmaxf(m, __shfl_xor(m, 1, 32));
            m = fmaxf(m, __shfl_xor(m, 2, 32));
            m = fmaxf(m, __shfl_xor(m, 4, 32));
            m = fmaxf(m, __shfl_xor(m, 8, 32));
            nmax[i] = m;
        }
        float scale[8];
        #pragma unroll
        for (int i = 0; i < 8; ++i) {
            scale[i] = __expf(rmax[i] - nmax[i]);
            rmax[i]  = nmax[i];
            rsum[i] *= scale[i];
        }
        #pragma unroll
        for (int t = 0; t < 4; ++t)
            #pragma unroll
            for (int i = 0; i < 8; ++i) oacc[t][i] *= scale[i];

        // ---- P = exp(s - max): row-sum partials, C->A layout via LDS ----
        #pragma unroll
        for (int ns = 0; ns < 4; ++ns) {
            #pragma unroll
            for (int i = 0; i < 8; ++i) {
                float p = __expf(sc[ns][i] - nmax[i]);
                rsum[i] += p;
                Ps[wid][i + 8 * lhal][ns * 16 + lrow] = (_Float16)p;
            }
        }
        v16h pa0 = load_frag_a(&Ps[wid][0][0],      72, lane);   // keys 0..31
        v16h pa1 = load_frag_a(&Ps[wid][0][0] + 32, 72, lane);   // keys 32..63

        // ---- O += P @ V ----
        #pragma unroll
        for (int dt = 0; dt < 4; ++dt) {
            v16h bv0 = load_frag_b(&Vt[dt * 16][0],      72, lane);
            v16h bv1 = load_frag_b(&Vt[dt * 16][0] + 32, 72, lane);
            oacc[dt] = wmma_f16(pa0, bv0, oacc[dt]);
            oacc[dt] = wmma_f16(pa1, bv1, oacc[dt]);
        }
        __syncthreads();
    }

    // ---- finalize: reduce row-sums, normalize, store ctx f16 [S][E] ----
    #pragma unroll
    for (int i = 0; i < 8; ++i) {
        float s = rsum[i];
        s += __shfl_xor(s, 1, 32);
        s += __shfl_xor(s, 2, 32);
        s += __shfl_xor(s, 4, 32);
        s += __shfl_xor(s, 8, 32);
        rsum[i] = 1.0f / s;
    }
    #pragma unroll
    for (int dt = 0; dt < 4; ++dt) {
        int d = dt * 16 + lrow;
        #pragma unroll
        for (int i = 0; i < 8; ++i) {
            int s = q0 + i + 8 * lhal;
            ctx[(size_t)s * E_DIM + h * HD + d] = (_Float16)(oacc[dt][i] * rsum[i]);
        }
    }
}

// ---------------------------------------------------------------------------
extern "C" void kernel_launch(void* const* d_in, const int* in_sizes, int n_in,
                              void* d_out, int out_size, void* d_ws, size_t ws_size,
                              hipStream_t stream) {
    (void)in_sizes; (void)n_in; (void)out_size;
    const float* x  = (const float*)d_in[0];
    const float* wq = (const float*)d_in[1];
    const float* bq = (const float*)d_in[2];
    const float* wk = (const float*)d_in[3];
    const float* bk = (const float*)d_in[4];
    const float* wv = (const float*)d_in[5];
    const float* bv = (const float*)d_in[6];
    const float* wo = (const float*)d_in[7];
    const float* bo = (const float*)d_in[8];
    float* out = (float*)d_out;

    const size_t nx = (size_t)S_LEN * E_DIM;   // 4M elems
    const size_t nw = (size_t)E_DIM * E_DIM;   // 1M elems
    if (ws_size < (nx * 5 + nw * 4) * sizeof(_Float16)) return;  // 48 MB

    _Float16* x16  = (_Float16*)d_ws;
    _Float16* wq16 = x16 + nx;
    _Float16* wk16 = wq16 + nw;
    _Float16* wv16 = wk16 + nw;
    _Float16* wo16 = wv16 + nw;
    _Float16* qh   = wo16 + nw;
    _Float16* kh   = qh + nx;
    _Float16* vh   = kh + nx;
    _Float16* ctx  = vh + nx;

    cvt_f16_kernel<<<(int)(nx / 2048), 256, 0, stream>>>(x,  x16,  (int)nx);
    cvt_f16_kernel<<<(int)(nw / 2048), 256, 0, stream>>>(wq, wq16, (int)nw);
    cvt_f16_kernel<<<(int)(nw / 2048), 256, 0, stream>>>(wk, wk16, (int)nw);
    cvt_f16_kernel<<<(int)(nw / 2048), 256, 0, stream>>>(wv, wv16, (int)nw);
    cvt_f16_kernel<<<(int)(nw / 2048), 256, 0, stream>>>(wo, wo16, (int)nw);

    dim3 gproj(E_DIM / 128, S_LEN / 128);              // (8, 32)
    gemm_wmma_kernel<0><<<gproj, 256, 0, stream>>>(x16, wq16, bq, qh, nullptr);
    gemm_wmma_kernel<0><<<gproj, 256, 0, stream>>>(x16, wk16, bk, kh, nullptr);
    gemm_wmma_kernel<0><<<gproj, 256, 0, stream>>>(x16, wv16, bv, vh, nullptr);
    attn_kernel<<<dim3(8, NWIN, HEADS), 128, 0, stream>>>(qh, kh, vh, ctx);
    gemm_wmma_kernel<1><<<gproj, 256, 0, stream>>>(ctx, wo16, bo, nullptr, out);
}
